// RGCNBlockLayer_48524540510309
// MI455X (gfx1250) — matrix operand order repaired and checked by reading, not compiled, not run on hardware
//
#include <hip/hip_runtime.h>
#include <hip/hip_bf16.h>

#define NNODES   50000
#define NEDGES   800000
#define DIM      128
#define NBASES   4
#define NTILES   (NNODES / 16)          // 3125 row tiles
#define NKSTEPS  (DIM / 32)             // 4 K-steps of 32

typedef __attribute__((ext_vector_type(16))) __bf16 v16bf;
typedef __attribute__((ext_vector_type(8)))  float  v8f;

// ---------------------------------------------------------------------------
// Pass 0: zero the aggregation buffer (re-zeroed every call).
// ---------------------------------------------------------------------------
__global__ __launch_bounds__(256)
void rgcn_zero(float* __restrict__ p, int n) {
    int i = blockIdx.x * 256 + threadIdx.x;
    if (i < n) p[i] = 0.0f;
}

// ---------------------------------------------------------------------------
// Pack the 5 weight matrices (4 bases + loop) into bf16, fragment-major:
// record index t = ((w*8 + colTile)*4 + kstep)*32 + lane, 16 bf16 per record.
// B-fragment layout (32x16 KxN, ISA 7.12.2): lane l -> N = l&15, half = l>>4;
// element j=2v+odd -> W[kstep*32 + (v/4)*16 + half*8 + (v%4)*2 + odd][colTile*16 + N].
// ---------------------------------------------------------------------------
__global__ __launch_bounds__(256)
void rgcn_pack_w(const float* __restrict__ basis,
                 const float* __restrict__ loopW,
                 __bf16* __restrict__ Wp)
{
    int t = blockIdx.x * 256 + threadIdx.x;
    if (t >= 5 * 8 * NKSTEPS * 32) return;
    const int l  = t & 31;
    const int ks = (t >> 5) & 3;
    const int c  = (t >> 7) & 7;
    const int w  = t >> 10;
    const float* __restrict__ W =
        (w < NBASES) ? (basis + (size_t)w * DIM * DIM) : loopW;
    const int hf = l >> 4, n = l & 15;

    v16bf rec;
    #pragma unroll
    for (int v = 0; v < 8; ++v) {
        const int k = ks * 32 + ((v >> 2) << 4) + (hf << 3) + ((v & 3) << 1);
        rec[2 * v]     = (__bf16)W[(size_t)k * DIM + c * 16 + n];
        rec[2 * v + 1] = (__bf16)W[(size_t)(k + 1) * DIM + c * 16 + n];
    }
    ((v16bf*)Wp)[t] = rec;
}

// ---------------------------------------------------------------------------
// Pack h into bf16, fragment-major: record t = (tile*4 + kstep)*32 + lane.
// A-fragment layout (16x32 MxK): lane l -> M = l&15, half = l>>4;
// element j=2v+odd -> h[tile*16 + M][kstep*32 + (v/4)*16 + half*8 + (v%4)*2 + odd].
// ---------------------------------------------------------------------------
__global__ __launch_bounds__(256)
void rgcn_pack_h(const float* __restrict__ h, __bf16* __restrict__ hp)
{
    int t = blockIdx.x * 256 + threadIdx.x;
    if (t >= NTILES * NKSTEPS * 32) return;
    const int l    = t & 31;
    const int ks   = (t >> 5) & 3;
    const int tile = t >> 7;
    const int hf = l >> 4, m = l & 15;
    const float* __restrict__ row = h + (size_t)(tile * 16 + m) * DIM + ks * 32;

    v16bf rec;
    #pragma unroll
    for (int v = 0; v < 8; ++v) {
        const int ka = ((v >> 2) << 4) + (hf << 3) + ((v & 3) << 1);
        rec[2 * v]     = (__bf16)row[ka];
        rec[2 * v + 1] = (__bf16)row[ka + 1];
    }
    ((v16bf*)hp)[t] = rec;
}

// ---------------------------------------------------------------------------
// Pass A: 5 fused GEMMs via WMMA bf16->f32 from pre-packed fragments.
// gridDim.x = row tiles, gridDim.y = 5. Block = 8 waves; wave w owns the
// 16x16 tile at cols [16w,16w+16). Inner loop: 2+2 b128 loads + 1 wmma.
// EXEC is uniformly all-ones (no divergence; 50000 % 16 == 0).
// ---------------------------------------------------------------------------
__global__ __launch_bounds__(256)
void rgcn_project(const __bf16* __restrict__ hp,
                  const __bf16* __restrict__ Wp,
                  float* __restrict__ hb,
                  float* __restrict__ loopMsg)
{
    const int wsel = blockIdx.y;                      // 0..3 basis, 4 = loop
    float* __restrict__ Out =
        (wsel < NBASES) ? (hb + (size_t)wsel * NNODES * DIM) : loopMsg;

    const int tile = blockIdx.x;
    const int wave = threadIdx.x >> 5;
    const int lane = threadIdx.x & 31;

    const v16bf* __restrict__ Abase =
        (const v16bf*)hp + (size_t)tile * NKSTEPS * 32 + lane;
    const v16bf* __restrict__ Bbase =
        (const v16bf*)Wp + (size_t)(wsel * 8 + wave) * NKSTEPS * 32 + lane;

    v8f acc = {};
    #pragma unroll
    for (int ks = 0; ks < NKSTEPS; ++ks) {
        const v16bf a = Abase[ks * 32];
        const v16bf b = Bbase[ks * 32];
        acc = __builtin_amdgcn_wmma_f32_16x16x32_bf16(
            false, a, false, b, (short)0, acc, false, false);
    }

    const int l15 = lane & 15;
    const int hf  = lane >> 4;
    const int col0 = wave * 16;
    #pragma unroll
    for (int v = 0; v < 8; ++v) {
        Out[(size_t)(tile * 16 + hf * 8 + v) * DIM + col0 + l15] = acc[v];
    }
}

// ---------------------------------------------------------------------------
// Pass B: edge gather / coefficient mix / scatter-add. One wave per edge;
// lane j owns the float4 at cols [4j,4j+4). hb (102.4 MB) + agg (25.6 MB)
// are L2-resident (192 MB L2); f32 adds use hardware global_atomic_add_f32.
// ---------------------------------------------------------------------------
__global__ __launch_bounds__(256)
void rgcn_edges(const float* __restrict__ hb,
                const int* __restrict__ src,
                const int* __restrict__ dst,
                const int* __restrict__ rel,
                const float* __restrict__ coeff,
                float* __restrict__ agg,
                int nEdges)
{
    const int edge = blockIdx.x * 8 + (threadIdx.x >> 5);
    if (edge >= nEdges) return;
    const int lane = threadIdx.x & 31;

    const int s = src[edge];
    const int d = dst[edge];
    const int r = rel[edge];

    const float4 c = ((const float4*)coeff)[r];      // (NUM_RELS, 4) rows

    const size_t plane = (size_t)NNODES * DIM;
    const size_t roff  = (size_t)s * DIM + lane * 4;

    const float4 m0 = *(const float4*)(hb + roff);
    const float4 m1 = *(const float4*)(hb + plane + roff);
    const float4 m2 = *(const float4*)(hb + 2 * plane + roff);
    const float4 m3 = *(const float4*)(hb + 3 * plane + roff);

    const float r0 = c.x * m0.x + c.y * m1.x + c.z * m2.x + c.w * m3.x;
    const float r1 = c.x * m0.y + c.y * m1.y + c.z * m2.y + c.w * m3.y;
    const float r2 = c.x * m0.z + c.y * m1.z + c.z * m2.z + c.w * m3.z;
    const float r3 = c.x * m0.w + c.y * m1.w + c.z * m2.w + c.w * m3.w;

    float* o = agg + (size_t)d * DIM + lane * 4;
    unsafeAtomicAdd(o + 0, r0);
    unsafeAtomicAdd(o + 1, r1);
    unsafeAtomicAdd(o + 2, r2);
    unsafeAtomicAdd(o + 3, r3);
}

// ---------------------------------------------------------------------------
// Pass C: out = relu(agg * norm + loop_message)  (loop_message lives in out)
// ---------------------------------------------------------------------------
__global__ __launch_bounds__(256)
void rgcn_final(const float* __restrict__ agg,
                const float* __restrict__ norm,
                float* __restrict__ out, int n)
{
    int i = blockIdx.x * 256 + threadIdx.x;
    if (i < n) {
        const float v = agg[i] * norm[i >> 7] + out[i];
        out[i] = v > 0.0f ? v : 0.0f;
    }
}

// ---------------------------------------------------------------------------
extern "C" void kernel_launch(void* const* d_in, const int* in_sizes, int n_in,
                              void* d_out, int out_size, void* d_ws, size_t ws_size,
                              hipStream_t stream) {
    const float* h     = (const float*)d_in[0];
    const float* norm  = (const float*)d_in[1];
    const int*   src   = (const int*)  d_in[2];
    const int*   dst   = (const int*)  d_in[3];
    const int*   rel   = (const int*)  d_in[4];
    const float* basis = (const float*)d_in[5];
    const float* coeff = (const float*)d_in[6];
    const float* loopW = (const float*)d_in[7];

    float* out = (float*)d_out;

    // Workspace layout (bytes):
    //   agg : NNODES*DIM f32                      = 25.6 MB
    //   hb  : NBASES*NNODES*DIM f32               = 102.4 MB
    //   hp  : NTILES*NKSTEPS*32*16 bf16           = 12.8 MB
    //   Wp  : 5*8*NKSTEPS*32*16 bf16              = 160 KB
    float*  agg = (float*)d_ws;
    float*  hb  = agg + (size_t)NNODES * DIM;
    __bf16* hp  = (__bf16*)(hb + (size_t)NBASES * NNODES * DIM);
    __bf16* Wp  = hp + (size_t)NTILES * NKSTEPS * 32 * 16;

    const int nOut = NNODES * DIM;

    rgcn_zero<<<(nOut + 255) / 256, 256, 0, stream>>>(agg, nOut);

    rgcn_pack_w<<<(5 * 8 * NKSTEPS * 32 + 255) / 256, 256, 0, stream>>>(
        basis, loopW, Wp);
    rgcn_pack_h<<<(NTILES * NKSTEPS * 32 + 255) / 256, 256, 0, stream>>>(h, hp);

    dim3 gGemm(NTILES, NBASES + 1);                  // 3125 x 5
    rgcn_project<<<gGemm, 256, 0, stream>>>(hp, Wp, hb, out);

    rgcn_edges<<<(NEDGES + 7) / 8, 256, 0, stream>>>(hb, src, dst, rel, coeff,
                                                     agg, NEDGES);

    rgcn_final<<<(nOut + 255) / 256, 256, 0, stream>>>(agg, norm, out, nOut);
}
